// HyenaLikeMix_71416716198234
// MI455X (gfx1250) — matrix operand order repaired and checked by reading, not compiled, not run on hardware
//
#include <hip/hip_runtime.h>

// ---------------------------------------------------------------------------
// HyenaLikeMix on MI455X (gfx1250, wave32, WMMA + async-LDS staging)
//   cvt  : x, w_in, w_out f32 -> bf16 (one-shot, out of the GEMM hot loop)
//   GEMM1: vg[8192,8192] = xb * w_in_b^T        (bf16 WMMA, f32 acc, bf16 out)
//   conv : y[8192,4096]  = silu(dwconv(v)+b)*silu(g)  (LDS-tiled)
//   GEMM2: out[8192,1024] = y * w_out_b^T       (bf16 WMMA, f32 acc, f32 out)
// GEMM staging: global_load_async_to_lds_b128 (ASYNCcnt) + double-buffered LDS.
// ---------------------------------------------------------------------------

#define D_MODEL 1024
#define D_INNER 4096
#define KCONV   127
#define BATCH   2
#define SEQLEN  4096
#define MROWS   (BATCH * SEQLEN)   // 8192

typedef __attribute__((ext_vector_type(16))) __bf16     v16bf;
typedef __attribute__((ext_vector_type(8)))  float      v8f;
typedef __attribute__((ext_vector_type(4)))  float      f4;
typedef __attribute__((ext_vector_type(4)))  unsigned   u4;

// ---- bf16 helpers (round-to-nearest-even) ---------------------------------
__device__ __forceinline__ unsigned short bfbits(float f) {
  union { float f; unsigned u; } x; x.f = f;
  unsigned r = x.u + 0x7FFFu + ((x.u >> 16) & 1u);
  return (unsigned short)(r >> 16);
}
__device__ __forceinline__ float bf2f(unsigned short h) {
  union { unsigned u; float f; } x; x.u = ((unsigned)h) << 16;
  return x.f;
}
__device__ __forceinline__ unsigned pk2(float lo, float hi) {
  return (unsigned)bfbits(lo) | ((unsigned)bfbits(hi) << 16);
}

union FragBF { v16bf v; unsigned u[8]; u4 q[2]; };

// ---- CDNA5 async copy: 16B global -> LDS, tracked by ASYNCcnt --------------
__device__ __forceinline__ void async_copy16(unsigned short* lds_dst, const unsigned short* gsrc) {
  unsigned lds_off = (unsigned)(uintptr_t)lds_dst;  // low 32 bits of generic = LDS offset
  asm volatile("global_load_async_to_lds_b128 %0, %1, off"
               :: "v"(lds_off), "v"(gsrc) : "memory");
}
__device__ __forceinline__ void async_wait0() {
  asm volatile("s_wait_asynccnt 0" ::: "memory");
}

// ---------------------------------------------------------------------------
// f32 -> bf16 conversion (8 elems/thread, b128 in / b128 out)
// ---------------------------------------------------------------------------
__global__ __launch_bounds__(256) void cvt_f32_bf16_kernel(
    const float* __restrict__ in, unsigned short* __restrict__ out, int n8)
{
  int i = blockIdx.x * 256 + threadIdx.x;
  if (i >= n8) return;
  const float* p = in + (size_t)i * 8;
  f4 a = *(const f4*)(p);
  f4 b = *(const f4*)(p + 4);
  u4 o;
  o[0] = pk2(a[0], a[1]); o[1] = pk2(a[2], a[3]);
  o[2] = pk2(b[0], b[1]); o[3] = pk2(b[2], b[3]);
  *(u4*)(out + (size_t)i * 8) = o;
}

// ---------------------------------------------------------------------------
// WMMA GEMM: C[M,N] = A[M,K] * Bt[N,K]^T, A/B bf16 row-major (K-contiguous).
// Block: 256 thr = 8 waves as 2(M) x 4(N); wave tile 64x64 (4x4 WMMA frags).
// Block tile: BM=128 x BN=256, BK=32. LDS double-buffered via async loads.
// ---------------------------------------------------------------------------
#define BM 128
#define BN 256
#define BK 32

template <int C_BF16>
__global__ __launch_bounds__(256) void wmma_gemm_kernel(
    const unsigned short* __restrict__ A, const unsigned short* __restrict__ B,
    void* __restrict__ Cp, int Kdim, int lda, int ldb, int ldc)
{
  __shared__ unsigned short sA[2][BM * BK];   // 2 x 8 KB
  __shared__ unsigned short sB[2][BN * BK];   // 2 x 16 KB

  const int tid  = threadIdx.x;
  const int lane = tid & 31;
  const int wave = tid >> 5;
  const int wm = wave & 1;          // 0..1 -> 64-row band
  const int wn = wave >> 1;         // 0..3 -> 64-col band
  const int row0 = blockIdx.x * BM;
  const int col0 = blockIdx.y * BN;

  // cooperative tile fetch: 16B per lane per pass (4 lanes per 32-elem row)
  const int lr = tid >> 2;          // 0..63 row index step
  const int lc = (tid & 3) * 8;     // 0,8,16,24

  v8f acc[4][4];
#pragma unroll
  for (int i = 0; i < 4; ++i)
#pragma unroll
    for (int j = 0; j < 4; ++j) acc[i][j] = (v8f)(0.0f);

  const int KT = Kdim / BK;

  // prologue: stage k-tile 0 into buffer 0
  {
#pragma unroll
    for (int p = 0; p < 2; ++p) {           // A: 128 rows
      int r = lr + p * 64;
      async_copy16(&sA[0][r * BK + lc], A + (size_t)(row0 + r) * lda + lc);
    }
#pragma unroll
    for (int p = 0; p < 4; ++p) {           // B: 256 rows
      int r = lr + p * 64;
      async_copy16(&sB[0][r * BK + lc], B + (size_t)(col0 + r) * ldb + lc);
    }
    async_wait0();
    __syncthreads();
  }

#pragma unroll 1
  for (int kt = 0; kt < KT; ++kt) {
    const int cur = kt & 1;
    const int nxt = cur ^ 1;

    // stage next k-tile while computing on current
    if (kt + 1 < KT) {
      const int k0 = (kt + 1) * BK;
#pragma unroll
      for (int p = 0; p < 2; ++p) {
        int r = lr + p * 64;
        async_copy16(&sA[nxt][r * BK + lc], A + (size_t)(row0 + r) * lda + k0 + lc);
      }
#pragma unroll
      for (int p = 0; p < 4; ++p) {
        int r = lr + p * 64;
        async_copy16(&sB[nxt][r * BK + lc], B + (size_t)(col0 + r) * ldb + k0 + lc);
      }
    }

    // fragments from LDS (ds_load_b128), ISA 7.12.2 bf16 layouts
    const int m  = lane & 15;
    const int ka = (lane & 16) ? 8 : 0;     // A: two 8-elem chunks at ka, ka+16
    const int kb = (lane & 16) ? 8 : 0;     // B: two 8-elem chunks at kb, kb+16

    v16bf a[4], b[4];
#pragma unroll
    for (int i = 0; i < 4; ++i) {
      const unsigned short* pa = &sA[cur][(wm * 64 + i * 16 + m) * BK + ka];
      FragBF r;
      r.q[0] = *(const u4*)(pa);
      r.q[1] = *(const u4*)(pa + 16);
      a[i] = r.v;
    }
#pragma unroll
    for (int j = 0; j < 4; ++j) {
      const unsigned short* pb = &sB[cur][(wn * 64 + j * 16 + m) * BK + kb];
      FragBF r;
      r.q[0] = *(const u4*)(pb);
      r.q[1] = *(const u4*)(pb + 16);
      b[j] = r.v;
    }

#pragma unroll
    for (int i = 0; i < 4; ++i)
#pragma unroll
      for (int j = 0; j < 4; ++j)
        acc[i][j] = __builtin_amdgcn_wmma_f32_16x16x32_bf16(
            false, a[i], false, b[j], (short)0, acc[i][j], false, false);

    async_wait0();      // my async writes for 'nxt' have landed
    __syncthreads();    // everyone done reading 'cur' + everyone's loads landed
  }

  // Epilogue: 16x16 f32 C layout — lane l: N=l&15, VGPR r holds M=r+(l>=16?8:0)
  const int n     = lane & 15;
  const int rbase = (lane & 16) ? 8 : 0;
#pragma unroll
  for (int i = 0; i < 4; ++i) {
#pragma unroll
    for (int j = 0; j < 4; ++j) {
#pragma unroll
      for (int r = 0; r < 8; ++r) {
        size_t idx = (size_t)(row0 + wm * 64 + i * 16 + rbase + r) * ldc
                   + (col0 + wn * 64 + j * 16 + n);
        float val = acc[i][j][r];
        if (C_BF16) ((unsigned short*)Cp)[idx] = bfbits(val);
        else        ((float*)Cp)[idx] = val;
      }
    }
  }
}

// ---------------------------------------------------------------------------
// Causal depthwise conv (K=127) + bias + silu(v)*silu(g), bf16 in / bf16 out.
// ---------------------------------------------------------------------------
__device__ __forceinline__ float silu_f(float x) {
  return x * (1.0f / (1.0f + __expf(-x)));
}

__global__ __launch_bounds__(256) void conv_gate_kernel(
    const unsigned short* __restrict__ vg,
    const float* __restrict__ w_dw,
    const float* __restrict__ b_dw,
    unsigned short* __restrict__ y)
{
  const int TL = 64, TC = 32, HALO = KCONV - 1;   // 126
  const int ROWS = TL + HALO;                     // 190
  const int PAD = 33;
  __shared__ unsigned short sv[ROWS * PAD];
  __shared__ float sw[TC][KCONV];
  __shared__ float sb[TC];

  int b  = blockIdx.z;
  int l0 = blockIdx.x * TL;
  int c0 = blockIdx.y * TC;

  for (int idx = threadIdx.x; idx < ROWS * TC; idx += 256) {
    int rr = idx >> 5;
    int cc = idx & 31;
    int l  = l0 - HALO + rr;
    unsigned short h = 0;
    if (l >= 0)
      h = vg[((size_t)b * SEQLEN + l) * (2 * D_INNER) + c0 + cc];
    sv[rr * PAD + cc] = h;
  }
  for (int idx = threadIdx.x; idx < TC * KCONV; idx += 256) {
    int cc = idx / KCONV;
    int t  = idx % KCONV;
    sw[cc][t] = w_dw[(size_t)(c0 + cc) * KCONV + t];
  }
  if (threadIdx.x < TC) sb[threadIdx.x] = b_dw[c0 + threadIdx.x];
  __syncthreads();

  int cc = threadIdx.x & 31;
  int lt = threadIdx.x >> 5;
#pragma unroll 1
  for (int ll = lt; ll < TL; ll += 8) {
    float acc = sb[cc];
#pragma unroll 1
    for (int t = 0; t < KCONV; ++t)
      acc += sw[cc][t] * bf2f(sv[(ll + t) * PAD + cc]);
    int l = l0 + ll;
    float g = bf2f(vg[((size_t)b * SEQLEN + l) * (2 * D_INNER) + D_INNER + c0 + cc]);
    float yv = silu_f(acc) * silu_f(g);
    y[((size_t)b * SEQLEN + l) * D_INNER + c0 + cc] = bfbits(yv);
  }
}

// ---------------------------------------------------------------------------
extern "C" void kernel_launch(void* const* d_in, const int* in_sizes, int n_in,
                              void* d_out, int out_size, void* d_ws, size_t ws_size,
                              hipStream_t stream) {
  const float* x     = (const float*)d_in[0];   // [2,4096,1024]
  const float* w_in  = (const float*)d_in[1];   // [8192,1024]
  const float* w_dw  = (const float*)d_in[2];   // [4096,127]
  const float* b_dw  = (const float*)d_in[3];   // [4096]
  const float* w_out = (const float*)d_in[4];   // [1024,4096]
  float* out = (float*)d_out;                   // [2,4096,1024]

  char* ws = (char*)d_ws;
  size_t off = 0;
  unsigned short* xb   = (unsigned short*)(ws + off); off += (size_t)MROWS * D_MODEL * 2;          // 16 MiB
  unsigned short* winb = (unsigned short*)(ws + off); off += (size_t)(2 * D_INNER) * D_MODEL * 2;  // 16 MiB
  unsigned short* woutb= (unsigned short*)(ws + off); off += (size_t)D_MODEL * D_INNER * 2;        //  8 MiB
  unsigned short* vg   = (unsigned short*)(ws + off); off += (size_t)MROWS * 2 * D_INNER * 2;      // 128 MiB
  unsigned short* y    = (unsigned short*)(ws + off);                                              // 64 MiB

  dim3 blk(256);

  // One-shot f32 -> bf16 conversions
  {
    int n8;
    n8 = (MROWS * D_MODEL) / 8;
    cvt_f32_bf16_kernel<<<dim3((n8 + 255) / 256), blk, 0, stream>>>(x, xb, n8);
    n8 = (2 * D_INNER * D_MODEL) / 8;
    cvt_f32_bf16_kernel<<<dim3((n8 + 255) / 256), blk, 0, stream>>>(w_in, winb, n8);
    n8 = (D_MODEL * D_INNER) / 8;
    cvt_f32_bf16_kernel<<<dim3((n8 + 255) / 256), blk, 0, stream>>>(w_out, woutb, n8);
  }

  // GEMM1: vg = xb * winb^T   (M=8192, N=8192, K=1024)
  wmma_gemm_kernel<1><<<dim3(MROWS / BM, (2 * D_INNER) / BN), blk, 0, stream>>>(
      xb, winb, (void*)vg, D_MODEL, D_MODEL, D_MODEL, 2 * D_INNER);

  // Conv + gating
  conv_gate_kernel<<<dim3(SEQLEN / 64, D_INNER / 32, BATCH), blk, 0, stream>>>(
      vg, w_dw, b_dw, y);

  // GEMM2: out = y * woutb^T  (M=8192, N=1024, K=4096)
  wmma_gemm_kernel<0><<<dim3(MROWS / BM, D_MODEL / BN), blk, 0, stream>>>(
      y, woutb, (void*)out, D_INNER, D_INNER, D_INNER, D_MODEL);
}
